// TransformerBlock_57011395887691
// MI455X (gfx1250) — matrix operand order repaired and checked by reading, not compiled, not run on hardware
//
#include <hip/hip_runtime.h>
#include <math.h>

typedef __bf16 bf16;
typedef __attribute__((ext_vector_type(16))) __bf16 v16bf;
typedef __attribute__((ext_vector_type(8)))  __bf16 v8bf;
typedef __attribute__((ext_vector_type(8)))  float   v8f;
typedef __attribute__((ext_vector_type(4)))  int     v4i;

#define DEV __device__ __forceinline__

#ifndef __has_builtin
#define __has_builtin(x) 0
#endif

#if defined(__HIP_DEVICE_COMPILE__) && __has_builtin(__builtin_amdgcn_global_load_async_to_lds_b128)
#define HAVE_ASYNC_LDS 1
#else
#define HAVE_ASYNC_LDS 0
#endif

static constexpr int Bb = 2, Ss = 2048, Dd = 1024, Hh = 16, DHh = 64, Ff = 4096;
static constexpr int Mm = Bb * Ss;  // 4096 rows

// ---------------------------------------------------------------------------
// WMMA fragment helpers (wave32, 16x16x32 bf16 -> f32 accum)
// A 16x32: lane m = lane&15; g = lane>>4; elem i -> k = ks + 8g + i (i<8),
//          k = ks + 16 + 8g + (i-8) (i>=8)   [contiguous 8+8 bf16 per lane]
// B 32x16: lane n = lane&15; elem i -> same k mapping as A  [contiguous 8+8 bf16 per lane]
// C/D 16x16 f32: vgpr v -> m = v + 8g; n = lane&15
// ---------------------------------------------------------------------------
DEV v16bf mk16(v8bf lo, v8bf hi) {
  v16bf r;
#pragma unroll
  for (int i = 0; i < 8; ++i) { r[i] = lo[i]; r[i + 8] = hi[i]; }
  return r;
}

DEV v16bf frag_a(const bf16* lds, int lda, int row, int g, int ks) {
  const bf16* p = lds + row * lda + ks + 8 * g;
  return mk16(*(const v8bf*)p, *(const v8bf*)(p + 16));
}

DEV v16bf frag_b(const bf16* lds, int ldb, int nrow, int g, int ks) {
  const bf16* p = lds + nrow * ldb + ks + 8 * g;
  return mk16(*(const v8bf*)p, *(const v8bf*)(p + 16));
}

DEV v8f wmma_bf16(v16bf a, v16bf b, v8f c) {
  return __builtin_amdgcn_wmma_f32_16x16x32_bf16(false, a, false, b, (short)0, c,
                                                 false, false);
}

// 16-byte global -> LDS copy: async (ASYNCcnt / bypasses VGPRs) if available
DEV void cp16(bf16* dst_lds, const bf16* src) {
#if HAVE_ASYNC_LDS
  typedef __attribute__((address_space(1))) v4i* gptr_t;
  typedef __attribute__((address_space(3))) v4i* lptr_t;
  __builtin_amdgcn_global_load_async_to_lds_b128((gptr_t)(void*)src,
                                                 (lptr_t)(void*)dst_lds, 0, 0);
#else
  *(v8bf*)dst_lds = *(const v8bf*)src;
#endif
}

DEV void wait_async0() {
#if HAVE_ASYNC_LDS
#if __has_builtin(__builtin_amdgcn_s_wait_asynccnt)
  __builtin_amdgcn_s_wait_asynccnt(0);
#else
  asm volatile("s_wait_asynccnt 0x0" ::: "memory");
#endif
#endif
}

// ---------------------------------------------------------------------------
// LayerNorm over D=1024, one row per 256-thread block, bf16 output
// ---------------------------------------------------------------------------
__global__ __launch_bounds__(256) void ln_bf16_kernel(
    const float* __restrict__ x, const float* __restrict__ gamma,
    const float* __restrict__ beta, bf16* __restrict__ out) {
  __shared__ float red[256];
  const int row = blockIdx.x, tid = threadIdx.x;
  const float* xr = x + (size_t)row * Dd;
  float v[4];
  float s = 0.f;
#pragma unroll
  for (int i = 0; i < 4; ++i) { v[i] = xr[tid + i * 256]; s += v[i]; }
  red[tid] = s;
  __syncthreads();
  for (int off = 128; off > 0; off >>= 1) {
    if (tid < off) red[tid] += red[tid + off];
    __syncthreads();
  }
  const float mu = red[0] * (1.0f / Dd);
  __syncthreads();
  float q = 0.f;
#pragma unroll
  for (int i = 0; i < 4; ++i) { const float d = v[i] - mu; q += d * d; }
  red[tid] = q;
  __syncthreads();
  for (int off = 128; off > 0; off >>= 1) {
    if (tid < off) red[tid] += red[tid + off];
    __syncthreads();
  }
  const float rstd = rsqrtf(red[0] * (1.0f / Dd) + 1e-5f);
#pragma unroll
  for (int i = 0; i < 4; ++i) {
    const int c = tid + i * 256;
    out[(size_t)row * Dd + c] = (bf16)((v[i] - mu) * rstd * gamma[c] + beta[c]);
  }
}

// ---------------------------------------------------------------------------
// Weight conversions (one-time): fp32 -> bf16, plain and transposed ([K,N]->[N,K])
// ---------------------------------------------------------------------------
__global__ void f32_to_bf16_kernel(const float* __restrict__ in,
                                   bf16* __restrict__ out, int n) {
  for (int i = blockIdx.x * blockDim.x + threadIdx.x; i < n;
       i += gridDim.x * blockDim.x)
    out[i] = (bf16)in[i];
}

__global__ void transpose_f32_to_bf16_kernel(const float* __restrict__ in,
                                             bf16* __restrict__ out, int R, int C) {
  // in[R][C] -> out[C][R]
  const int total = R * C;
  for (int i = blockIdx.x * blockDim.x + threadIdx.x; i < total;
       i += gridDim.x * blockDim.x) {
    const int c = i / R;
    const int r = i - c * R;
    out[(size_t)c * R + r] = (bf16)in[(size_t)r * C + c];
  }
}

// ---------------------------------------------------------------------------
// Tiled bf16 WMMA GEMM: out[M,N] = A[M,K] * W[N,K]^T, with epilogue modes
//  1 = scatter Q/K/V bf16 to per-head contiguous [B*H, S, DH]
//  2 = fp32 out + residual
//  3 = bias + exact GELU -> bf16 out
//  4 = bias + residual -> fp32 out (final)
// Block tile 128x128x64, 8 waves (2x4), wave tile 64x32.
// Double-buffered LDS, async global->LDS staging pipelined against WMMA.
// ---------------------------------------------------------------------------
__global__ __launch_bounds__(256) void gemm_bf16_kernel(
    const bf16* __restrict__ A, const bf16* __restrict__ W, int M, int N, int K,
    int mode, float* __restrict__ outF, bf16* __restrict__ outB,
    const float* __restrict__ res, const float* __restrict__ bias,
    bf16* __restrict__ qh, bf16* __restrict__ kh, bf16* __restrict__ vh) {
  constexpr int BM = 128, BN = 128, BK = 64, LDT = 72;
  __shared__ alignas(16) bf16 As[2][BM * LDT];
  __shared__ alignas(16) bf16 Bs[2][BN * LDT];
  const int tid = threadIdx.x, wid = tid >> 5, lane = tid & 31;
  const int l16 = lane & 15, g = lane >> 4;
  const int n0 = blockIdx.x * BN, m0 = blockIdx.y * BM;
  const int wm = wid & 1, wn = wid >> 1;

  auto stage = [&](int buf, int kt) {
    for (int c = tid; c < BM * BK / 8; c += 256) {
      const int r = c >> 3, col = (c & 7) << 3;
      cp16(&As[buf][r * LDT + col], A + (size_t)(m0 + r) * K + kt + col);
    }
    for (int c = tid; c < BN * BK / 8; c += 256) {
      const int r = c >> 3, col = (c & 7) << 3;
      cp16(&Bs[buf][r * LDT + col], W + (size_t)(n0 + r) * K + kt + col);
    }
  };

  v8f acc[4][2];
#pragma unroll
  for (int i = 0; i < 4; ++i)
#pragma unroll
    for (int j = 0; j < 2; ++j)
#pragma unroll
      for (int v = 0; v < 8; ++v) acc[i][j][v] = 0.0f;

  stage(0, 0);  // prologue: fill buffer 0
  int buf = 0;
  for (int kt = 0; kt < K; kt += BK) {
    wait_async0();    // this wave's copies into `buf` are done
    __syncthreads();  // all waves' copies visible; prior reads of buf^1 retired
    if (kt + BK < K) stage(buf ^ 1, kt + BK);  // issue next tile (overlapped)
#pragma unroll
    for (int ks = 0; ks < BK; ks += 32) {
      v16bf af[4], bfr[2];
#pragma unroll
      for (int i = 0; i < 4; ++i)
        af[i] = frag_a(As[buf], LDT, wm * 64 + i * 16 + l16, g, ks);
#pragma unroll
      for (int j = 0; j < 2; ++j)
        bfr[j] = frag_b(Bs[buf], LDT, wn * 32 + j * 16 + l16, g, ks);
#pragma unroll
      for (int i = 0; i < 4; ++i)
#pragma unroll
        for (int j = 0; j < 2; ++j) acc[i][j] = wmma_bf16(af[i], bfr[j], acc[i][j]);
    }
    buf ^= 1;
  }

#pragma unroll
  for (int i = 0; i < 4; ++i)
#pragma unroll
    for (int j = 0; j < 2; ++j) {
      const int nc = n0 + wn * 32 + j * 16 + l16;
#pragma unroll
      for (int v = 0; v < 8; ++v) {
        const int mr = m0 + wm * 64 + i * 16 + v + 8 * g;
        const float val = acc[i][j][v];
        if (mode == 1) {
          const int part = nc >> 10, nn = nc & 1023;
          const int hh = nn >> 6, e = nn & 63;
          const int bb = mr >> 11, s = mr & 2047;
          bf16* dst = (part == 0) ? qh : ((part == 1) ? kh : vh);
          dst[((size_t)(bb * Hh + hh) * Ss + s) * DHh + e] = (bf16)val;
        } else if (mode == 2) {
          const size_t o = (size_t)mr * N + nc;
          outF[o] = val + res[o];
        } else if (mode == 3) {
          const float t = val + bias[nc];
          const float gl = 0.5f * t * (1.0f + erff(t * 0.70710678118654752f));
          outB[(size_t)mr * N + nc] = (bf16)gl;
        } else {
          const size_t o = (size_t)mr * N + nc;
          outF[o] = val + bias[nc] + res[o];
        }
      }
    }
}

// ---------------------------------------------------------------------------
// Flash attention with the reference quirks: scores = (Q @ reshape(K)) / D,
// reshape(K) per (b,h) is just the flat [S*DH] buffer viewed as [DH, S].
// Grid: (S/128 query tiles, B*H). 8 waves, 16 query rows each, 64-key chunks.
// ---------------------------------------------------------------------------
__global__ __launch_bounds__(256) void attn_kernel(
    const bf16* __restrict__ qh, const bf16* __restrict__ kh,
    const bf16* __restrict__ vh, bf16* __restrict__ hc) {
  constexpr int LDT = 72;
  __shared__ alignas(16) bf16 Kl[64 * LDT];      // [s' local][e']
  __shared__ alignas(16) bf16 Vl[64 * LDT];      // [e][s' local]
  __shared__ alignas(16) bf16 Pl[8 * 16 * LDT];  // per-wave P tile [m][s' local]
  const int tid = threadIdx.x, wid = tid >> 5, lane = tid & 31;
  const int l16 = lane & 15, g = lane >> 4;
  const int bh = blockIdx.y;
  const int b = bh >> 4, h = bh & 15;
  const int q0 = blockIdx.x * 128;
  const bf16* qb = qh + (size_t)bh * Ss * DHh;
  const bf16* kb = kh + (size_t)bh * Ss * DHh;  // flat [S*DH] == K_r [DH][S]
  const bf16* vb = vh + (size_t)bh * Ss * DHh;
  bf16* Pw = Pl + wid * 16 * LDT;

  // Q fragments (16 rows x 64 k) straight from global, A-fragment layout
  const int qrow = q0 + wid * 16 + l16;
  v16bf qf[2];
#pragma unroll
  for (int f = 0; f < 2; ++f) {
    const bf16* p = qb + (size_t)qrow * DHh + f * 32 + 8 * g;
    qf[f] = mk16(*(const v8bf*)p, *(const v8bf*)(p + 16));
  }

  v8f acc[4];
  float rmax[8], rsum[8];
#pragma unroll
  for (int e = 0; e < 4; ++e)
#pragma unroll
    for (int v = 0; v < 8; ++v) acc[e][v] = 0.0f;
#pragma unroll
  for (int v = 0; v < 8; ++v) { rmax[v] = -1e30f; rsum[v] = 0.0f; }

  for (int c0 = 0; c0 < Ss; c0 += 64) {
    __syncthreads();
    {  // stage K chunk transposed: Kl[s'][e'] = K_r[e'][c0+s']
      const int e = tid >> 2, sb = (tid & 3) * 16;
      const bf16* src = kb + (size_t)e * Ss + c0 + sb;
      const v8bf k0 = *(const v8bf*)src, k1 = *(const v8bf*)(src + 8);
#pragma unroll
      for (int j = 0; j < 8; ++j) {
        Kl[(sb + j) * LDT + e] = k0[j];
        Kl[(sb + 8 + j) * LDT + e] = k1[j];
      }
      // stage V chunk transposed: Vl[e][s'] = V[c0+s'][e]
      const int sv = tid >> 2, eb = (tid & 3) * 16;
      const bf16* vs = vb + (size_t)(c0 + sv) * DHh + eb;
      const v8bf w0 = *(const v8bf*)vs, w1 = *(const v8bf*)(vs + 8);
#pragma unroll
      for (int j = 0; j < 8; ++j) {
        Vl[(eb + j) * LDT + sv] = w0[j];
        Vl[(eb + 8 + j) * LDT + sv] = w1[j];
      }
    }
    __syncthreads();

    v8f sc[4];
#pragma unroll
    for (int nt = 0; nt < 4; ++nt) {
      v8f z;
#pragma unroll
      for (int v = 0; v < 8; ++v) z[v] = 0.0f;
#pragma unroll
      for (int ks = 0; ks < 64; ks += 32)
        z = wmma_bf16(qf[ks >> 5], frag_b(Kl, LDT, nt * 16 + l16, g, ks), z);
      sc[nt] = z;
    }

    constexpr float INV_D = 1.0f / 1024.0f;  // reference quirk: /d_model
#pragma unroll
    for (int v = 0; v < 8; ++v) {
      const float s0 = sc[0][v] * INV_D, s1 = sc[1][v] * INV_D;
      const float s2 = sc[2][v] * INV_D, s3 = sc[3][v] * INV_D;
      float mx = fmaxf(fmaxf(s0, s1), fmaxf(s2, s3));
#pragma unroll
      for (int off = 1; off < 16; off <<= 1) mx = fmaxf(mx, __shfl_xor(mx, off, 32));
      const float mnew = fmaxf(rmax[v], mx);
      const float corr = __expf(rmax[v] - mnew);
      const float p0 = __expf(s0 - mnew), p1 = __expf(s1 - mnew);
      const float p2 = __expf(s2 - mnew), p3 = __expf(s3 - mnew);
      const int prow = (v + 8 * g) * LDT + l16;
      Pw[prow] = (bf16)p0;
      Pw[prow + 16] = (bf16)p1;
      Pw[prow + 32] = (bf16)p2;
      Pw[prow + 48] = (bf16)p3;
      float psum = p0 + p1 + p2 + p3;
#pragma unroll
      for (int off = 1; off < 16; off <<= 1) psum += __shfl_xor(psum, off, 32);
      rsum[v] = rsum[v] * corr + psum;
      rmax[v] = mnew;
#pragma unroll
      for (int e = 0; e < 4; ++e) acc[e][v] *= corr;
    }

#pragma unroll
    for (int et = 0; et < 4; ++et)
#pragma unroll
      for (int ks = 0; ks < 64; ks += 32)
        acc[et] = wmma_bf16(frag_a(Pw, LDT, l16, g, ks),
                            frag_b(Vl, LDT, et * 16 + l16, g, ks), acc[et]);
  }

  // heads_cat[b, s, h*64 + e] (transpose(0,2,1,3).reshape)
#pragma unroll
  for (int et = 0; et < 4; ++et)
#pragma unroll
    for (int v = 0; v < 8; ++v) {
      const int s = q0 + wid * 16 + v + 8 * g;
      const int col = h * DHh + et * 16 + l16;
      hc[(size_t)(b * Ss + s) * (Hh * DHh) + col] = (bf16)(acc[et][v] / rsum[v]);
    }
}

// ---------------------------------------------------------------------------
extern "C" void kernel_launch(void* const* d_in, const int* in_sizes, int n_in,
                              void* d_out, int out_size, void* d_ws, size_t ws_size,
                              hipStream_t stream) {
  const float* x      = (const float*)d_in[0];
  const float* W_Q    = (const float*)d_in[1];
  const float* W_K    = (const float*)d_in[2];
  const float* W_V    = (const float*)d_in[3];
  const float* W_O    = (const float*)d_in[4];
  const float* W_up   = (const float*)d_in[5];
  const float* b_up   = (const float*)d_in[6];
  const float* W_down = (const float*)d_in[7];
  const float* b_down = (const float*)d_in[8];
  const float* gamma1 = (const float*)d_in[9];
  const float* beta1  = (const float*)d_in[10];
  const float* gamma2 = (const float*)d_in[11];
  const float* beta2  = (const float*)d_in[12];
  float* out = (float*)d_out;

  char* ws = (char*)d_ws;
  size_t off = 0;
  auto alloc = [&](size_t bytes) -> char* {
    char* p = ws + off;
    off += (bytes + 255) & ~(size_t)255;
    return p;
  };
  bf16* u_bf  = (bf16*)alloc((size_t)Mm * Dd * 2);      // LN1 out
  bf16* Wqkv  = (bf16*)alloc((size_t)3 * Dd * Dd * 2);  // [3072][1024] (N,K)
  bf16* Wo_t  = (bf16*)alloc((size_t)Dd * Dd * 2);      // [1024][1024] (N,K)
  bf16* Wup_t = (bf16*)alloc((size_t)Ff * Dd * 2);      // [4096][1024] (N,K)
  bf16* Wdn_t = (bf16*)alloc((size_t)Dd * Ff * 2);      // [1024][4096] (N,K)
  bf16* qhB   = (bf16*)alloc((size_t)Mm * Dd * 2);      // [B*H][S][DH]
  bf16* khB   = (bf16*)alloc((size_t)Mm * Dd * 2);
  bf16* vhB   = (bf16*)alloc((size_t)Mm * Dd * 2);
  bf16* hc    = (bf16*)alloc((size_t)Mm * Dd * 2);      // heads_cat
  float* z1   = (float*)alloc((size_t)Mm * Dd * 4);
  bf16* v1_bf = (bf16*)alloc((size_t)Mm * Dd * 2);
  bf16* act   = (bf16*)alloc((size_t)Mm * Ff * 2);

  ln_bf16_kernel<<<Mm, 256, 0, stream>>>(x, gamma1, beta1, u_bf);
  f32_to_bf16_kernel<<<1024, 256, 0, stream>>>(W_Q, Wqkv, Dd * Dd);
  f32_to_bf16_kernel<<<1024, 256, 0, stream>>>(W_K, Wqkv + (size_t)Dd * Dd, Dd * Dd);
  f32_to_bf16_kernel<<<1024, 256, 0, stream>>>(W_V, Wqkv + (size_t)2 * Dd * Dd, Dd * Dd);
  transpose_f32_to_bf16_kernel<<<1024, 256, 0, stream>>>(W_O, Wo_t, Dd, Dd);
  transpose_f32_to_bf16_kernel<<<2048, 256, 0, stream>>>(W_up, Wup_t, Dd, Ff);
  transpose_f32_to_bf16_kernel<<<2048, 256, 0, stream>>>(W_down, Wdn_t, Ff, Dd);

  // QKV projections: [4096,1024] x [3072,1024]^T, scatter to per-head buffers
  gemm_bf16_kernel<<<dim3(3 * Dd / 128, Mm / 128), 256, 0, stream>>>(
      u_bf, Wqkv, Mm, 3 * Dd, Dd, 1, nullptr, nullptr, nullptr, nullptr, qhB, khB, vhB);
  // Flash attention -> heads_cat
  attn_kernel<<<dim3(Ss / 128, Bb * Hh), 256, 0, stream>>>(qhB, khB, vhB, hc);
  // z1 = x + heads_cat @ W_O
  gemm_bf16_kernel<<<dim3(Dd / 128, Mm / 128), 256, 0, stream>>>(
      hc, Wo_t, Mm, Dd, Dd, 2, z1, nullptr, x, nullptr, nullptr, nullptr, nullptr);
  ln_bf16_kernel<<<Mm, 256, 0, stream>>>(z1, gamma2, beta2, v1_bf);
  // act = gelu(v1 @ W_up + b_up)
  gemm_bf16_kernel<<<dim3(Ff / 128, Mm / 128), 256, 0, stream>>>(
      v1_bf, Wup_t, Mm, Ff, Dd, 3, nullptr, act, nullptr, b_up, nullptr, nullptr, nullptr);
  // out = z1 + act @ W_down + b_down
  gemm_bf16_kernel<<<dim3(Dd / 128, Mm / 128), 256, 0, stream>>>(
      act, Wdn_t, Mm, Dd, Ff, 4, out, nullptr, z1, b_down, nullptr, nullptr, nullptr);

  (void)in_sizes; (void)n_in; (void)out_size; (void)ws_size;
}
